// SingleSelfAttention_51831665328428
// MI455X (gfx1250) — matrix operand order repaired and checked by reading, not compiled, not run on hardware
//
#include <hip/hip_runtime.h>
#include <hip/hip_bf16.h>
#include <cstdint>
#include <cstddef>

// ---------------------------------------------------------------------------
// Self-attention (B=4, N=2048, hidden=1024, 16 heads x 64) for gfx1250.
//  - all GEMMs on V_WMMA_F32_16X16X32_BF16 (f32 accumulate)
//  - weight / K / V tiles staged into LDS with GLOBAL_LOAD_ASYNC_TO_LDS_B128
//    (ASYNCcnt), double-buffered behind the WMMA loop, shared by 8 waves
//  - flash-style online softmax in f32
// ---------------------------------------------------------------------------

typedef __bf16 bf16;
typedef __bf16 bf16x4  __attribute__((ext_vector_type(4)));
typedef __bf16 bf16x8  __attribute__((ext_vector_type(8)));
typedef __bf16 bf16x16 __attribute__((ext_vector_type(16)));
typedef float  f32x8   __attribute__((ext_vector_type(8)));

#define HIDDEN   1024
#define HEADS    16
#define HEAD_DIM 64
#define SEQ      2048
#define BATCH    4
#define TOKENS   (BATCH * SEQ)   // 8192

// Async copy global -> LDS, 16B per lane.  VDST VGPR holds the per-lane LDS
// byte address (low 32 bits of the flat pointer == group-segment offset).
#define ASYNC_B128(ldsptr, gptr)                                           \
  asm volatile("global_load_async_to_lds_b128 %0, %1, off"                 \
               :: "v"((uint32_t)(uintptr_t)(ldsptr)),                      \
                  "v"((uint64_t)(uintptr_t)(gptr))                         \
               : "memory")

#define WAIT_ASYNC(n) asm volatile("s_wait_asynccnt " #n ::: "memory")
#define WAIT_DS0()    asm volatile("s_wait_dscnt 0x0" ::: "memory")

static __device__ __forceinline__ bf16x16 cat16(bf16x8 lo, bf16x8 hi) {
  return __builtin_shufflevector(lo, hi, 0,1,2,3,4,5,6,7,8,9,10,11,12,13,14,15);
}

// A fragment (16x32 bf16, M x K) from row-major src (leading dim ld).
// ISA layout: lane = {m, half}; half 0 holds k = {0..7, 16..23},
// half 1 holds k = {8..15, 24..31} -> two b128 loads.
static __device__ __forceinline__ bf16x16
load_a_frag(const bf16* __restrict__ src, int ld, int row0, int col0, int lane) {
  const int mrow = lane & 15;
  const int hh   = lane >> 4;
  const bf16* p = src + (size_t)(row0 + mrow) * ld + col0 + hh * 8;
  bf16x8 lo = *(const bf16x8*)p;
  bf16x8 hi = *(const bf16x8*)(p + 16);
  return cat16(lo, hi);
}

// B fragment (32x16 bf16, K x N) given B-transposed row-major bt[N][K]
// (leading dim ld): lane = {n, half}; half h holds k = {8h..8h+7, 16+8h..16+8h+7}.
// Works for global or LDS-resident bt (addrspace inferred after inlining).
static __device__ __forceinline__ bf16x16
load_bt_frag(const bf16* __restrict__ bt, int ld, int n0, int k0, int lane) {
  const int n  = lane & 15;
  const int hh = lane >> 4;
  const bf16* p = bt + (size_t)(n0 + n) * ld + k0 + hh * 8;
  bf16x8 lo = *(const bf16x8*)p;
  bf16x8 hi = *(const bf16x8*)(p + 16);
  return cat16(lo, hi);
}

static __device__ __forceinline__ f32x8 wmma_bf16(bf16x16 a, bf16x16 b, f32x8 c) {
  return __builtin_amdgcn_wmma_f32_16x16x32_bf16(false, a, false, b, (short)0, c,
                                                 false, false);
}

// ---------------------------------------------------------------------------
// Kernel 0: f32 -> bf16 conversion, 4 elements/thread.
// ---------------------------------------------------------------------------
__global__ void cvt_f32_to_bf16_v4(const float* __restrict__ in,
                                   bf16* __restrict__ out, int n4) {
  int i = blockIdx.x * blockDim.x + threadIdx.x;
  if (i < n4) {
    float4 v = ((const float4*)in)[i];
    bf16x4 o;
    o[0] = (bf16)v.x; o[1] = (bf16)v.y; o[2] = (bf16)v.z; o[3] = (bf16)v.w;
    ((bf16x4*)out)[i] = o;
  }
}

// ---------------------------------------------------------------------------
// Kernel 1: QKV GEMM.  C[8192,3072] = X[8192,1024] * Wqkv^T.
// 256-thread block = 8 waves; block tile 256(M) x 64(N).
// Weight tile Bt[64][32] (4 KB) async-staged to LDS (1 async inst/wave),
// double-buffered; each wave: 2 A frags (global) + 4 shared B frags (LDS)
// -> 8 WMMAs / K-step.  Outputs scattered into q[bh][y][d] (pre-scaled),
// k[bh][y][d], vT[bh][d][y].
// ---------------------------------------------------------------------------
__global__ __launch_bounds__(256)
void qkv_gemm_kernel(const bf16* __restrict__ xb, const bf16* __restrict__ wq,
                     bf16* __restrict__ qo, bf16* __restrict__ ko,
                     bf16* __restrict__ vto) {
  __shared__ __align__(16) bf16 wbuf[2][64][32];

  const int tid  = threadIdx.x;
  const int lane = tid & 31;
  const int w    = tid >> 5;
  const int rowA = blockIdx.x * 256 + w * 32;  // this wave's 32 M rows
  const int col0 = blockIdx.y * 64;

  auto stage = [&](int kk, int bufi) {        // 1 async b128 inst / wave
    const int n = tid >> 2, off = (tid & 3) * 8;   // 256 chunks of 16B
    ASYNC_B128(&wbuf[bufi][n][off],
               wq + (size_t)(col0 + n) * HIDDEN + kk + off);
  };

  f32x8 acc[2][4] = {};
  stage(0, 0);
  stage(32, 1);

  const int NSTEP = HIDDEN / 32;              // 32
  for (int s = 0; s < NSTEP; ++s) {
    if (s + 1 < NSTEP) { WAIT_ASYNC(0x1); } else { WAIT_ASYNC(0x0); }
    __syncthreads();
    const int bufi = s & 1;
    const int kk   = s * 32;
    bf16x16 a0 = load_a_frag(xb, HIDDEN, rowA,      kk, lane);
    bf16x16 a1 = load_a_frag(xb, HIDDEN, rowA + 16, kk, lane);
    const bf16* wb = &wbuf[bufi][0][0];
#pragma unroll
    for (int t = 0; t < 4; ++t) {
      bf16x16 b = load_bt_frag(wb, 32, t * 16, 0, lane);
      acc[0][t] = wmma_bf16(a0, b, acc[0][t]);
      acc[1][t] = wmma_bf16(a1, b, acc[1][t]);
    }
    __syncthreads();
    if (s + 2 < NSTEP) {
      stage(kk + 64, bufi);
      // WGP-scope prefetch of this wave's next-next A chunk.
      __builtin_prefetch(xb + (size_t)(rowA + lane) * HIDDEN + kk + 64, 0, 3);
    }
  }

  const float qscale = 0.125f;                // HEAD_DIM^-0.5 folded into Q
  const int nloc = lane & 15;
  const int hh   = lane >> 4;
#pragma unroll
  for (int g = 0; g < 2; ++g) {
    const int rbase = rowA + g * 16;
#pragma unroll
    for (int t = 0; t < 4; ++t) {
      const int n     = col0 + t * 16 + nloc; // output feature in [0,3072)
      const int which = n >> 10;              // 0 = Q, 1 = K, 2 = V
      const int f     = n & (HIDDEN - 1);
      const int hd    = f >> 6;
      const int d     = f & (HEAD_DIM - 1);
#pragma unroll
      for (int r = 0; r < 8; ++r) {
        const int tok = rbase + r + 8 * hh;
        const int b_  = tok >> 11;
        const int y   = tok & (SEQ - 1);
        const size_t bh = (size_t)b_ * HEADS + hd;
        const float v = acc[g][t][r];
        if (which == 0)
          qo[(bh * SEQ + y) * HEAD_DIM + d] = (bf16)(v * qscale);
        else if (which == 1)
          ko[(bh * SEQ + y) * HEAD_DIM + d] = (bf16)v;
        else
          vto[(bh * HEAD_DIM + d) * SEQ + y] = (bf16)v;
      }
    }
  }
}

// ---------------------------------------------------------------------------
// Kernel 2: flash attention.  256-thread block = 8 waves; block owns 128
// query rows of one (b,h), each wave 16 rows.  The 32-key K tile [32][64]
// and V^T tile [64][32] are async-staged to LDS (double-buffered) and shared
// by all 8 waves.  S = Q*K^T (4 WMMAs), f32 online softmax, P through
// wave-local LDS, O += P*V (4 WMMAs).
// ---------------------------------------------------------------------------
__global__ __launch_bounds__(256)
void attn_kernel(const bf16* __restrict__ q, const bf16* __restrict__ k,
                 const bf16* __restrict__ vt, bf16* __restrict__ attnout) {
  __shared__ __align__(16) bf16 kbuf[2][32][64];   // 2 x 4 KB
  __shared__ __align__(16) bf16 vbuf[2][64][32];   // 2 x 4 KB
  __shared__ __align__(16) bf16 psh[8][16][32];    // 8 x 1 KB (per wave)

  const int tid  = threadIdx.x;
  const int lane = tid & 31;
  const int w    = tid >> 5;
  const int row0 = blockIdx.x * 128 + w * 16;      // this wave's query rows
  const int bh   = blockIdx.y;
  const int b_   = bh >> 4;
  const int h    = bh & 15;

  const bf16* qp = q  + (size_t)bh * SEQ * HEAD_DIM;
  const bf16* kp = k  + (size_t)bh * SEQ * HEAD_DIM;
  const bf16* vp = vt + (size_t)bh * HEAD_DIM * SEQ;

  auto stage_tile = [&](int y0, int bufi) {        // 2 async insts / wave
    {  // K tile rows [y0, y0+32), layout [32][64]: 256 chunks of 16B
      const int y = tid >> 3, off = (tid & 7) * 8;
      ASYNC_B128(&kbuf[bufi][y][off], kp + (size_t)(y0 + y) * HEAD_DIM + off);
    }
    {  // V^T tile cols [y0, y0+32), layout [64][32]: 256 chunks of 16B
      const int d = tid >> 2, off = (tid & 3) * 8;
      ASYNC_B128(&vbuf[bufi][d][off], vp + (size_t)d * SEQ + y0 + off);
    }
  };

  const bf16x16 a0 = load_a_frag(qp, HEAD_DIM, row0, 0, lane);
  const bf16x16 a1 = load_a_frag(qp, HEAD_DIM, row0, 32, lane);

  f32x8 o[4] = {};
  float mrun[8], lrun[8];
#pragma unroll
  for (int r = 0; r < 8; ++r) { mrun[r] = -1e30f; lrun[r] = 0.0f; }

  const int nloc = lane & 15;
  const int hh   = lane >> 4;

  stage_tile(0, 0);
  stage_tile(32, 1);

  const int NSTEP = SEQ / 32;                      // 64
  for (int s = 0; s < NSTEP; ++s) {
    if (s + 1 < NSTEP) { WAIT_ASYNC(0x2); } else { WAIT_ASYNC(0x0); }
    __syncthreads();
    const int bufi = s & 1;
    const bf16* kb = &kbuf[bufi][0][0];
    const bf16* vb = &vbuf[bufi][0][0];

    // S tile: 16 rows x 32 keys = two C fragments (B frags from LDS).
    f32x8 s0 = {}, s1 = {};
    s0 = wmma_bf16(a0, load_bt_frag(kb, 64, 0,  0,  lane), s0);
    s0 = wmma_bf16(a1, load_bt_frag(kb, 64, 0,  32, lane), s0);
    s1 = wmma_bf16(a0, load_bt_frag(kb, 64, 16, 0,  lane), s1);
    s1 = wmma_bf16(a1, load_bt_frag(kb, 64, 16, 32, lane), s1);

    // Online softmax update; stash P (bf16) in wave-local LDS.
#pragma unroll
    for (int r = 0; r < 8; ++r) {
      float v0 = s0[r], v1 = s1[r];
      float tm = fmaxf(v0, v1);
      tm = fmaxf(tm, __shfl_xor(tm, 1, 32));
      tm = fmaxf(tm, __shfl_xor(tm, 2, 32));
      tm = fmaxf(tm, __shfl_xor(tm, 4, 32));
      tm = fmaxf(tm, __shfl_xor(tm, 8, 32));
      const float mnew = fmaxf(mrun[r], tm);
      const float corr = __expf(mrun[r] - mnew);
      const float p0 = __expf(v0 - mnew);
      const float p1 = __expf(v1 - mnew);
      float rs = p0 + p1;
      rs += __shfl_xor(rs, 1, 32);
      rs += __shfl_xor(rs, 2, 32);
      rs += __shfl_xor(rs, 4, 32);
      rs += __shfl_xor(rs, 8, 32);
      lrun[r] = lrun[r] * corr + rs;
      mrun[r] = mnew;
#pragma unroll
      for (int t = 0; t < 4; ++t) o[t][r] *= corr;
      const int rowm = r + 8 * hh;
      psh[w][rowm][nloc]      = (bf16)p0;
      psh[w][rowm][16 + nloc] = (bf16)p1;
    }
    WAIT_DS0();   // wave-local DS ordering before re-reading P

    // Re-load P as A fragment (16x32) from LDS.
    const bf16* pp = &psh[w][lane & 15][hh * 8];
    bf16x8 plo = *(const bf16x8*)pp;
    bf16x8 phi = *(const bf16x8*)(pp + 16);
    bf16x16 pa = cat16(plo, phi);

    // O += P * V  (B frags from LDS vbuf = Bt[N=d][K=y]).
#pragma unroll
    for (int t = 0; t < 4; ++t)
      o[t] = wmma_bf16(pa, load_bt_frag(vb, 32, t * 16, 0, lane), o[t]);

    __syncthreads();
    if (s + 2 < NSTEP) stage_tile((s + 2) * 32, bufi);
  }

  // Normalize and write attn output (bf16) at [b, y, h*64 + d].
#pragma unroll
  for (int t = 0; t < 4; ++t) {
    const int d = t * 16 + nloc;
#pragma unroll
    for (int r = 0; r < 8; ++r) {
      const int y = row0 + r + 8 * hh;
      const size_t tok = (size_t)b_ * SEQ + y;
      attnout[tok * HIDDEN + h * HEAD_DIM + d] = (bf16)(o[t][r] / lrun[r]);
    }
  }
}

// ---------------------------------------------------------------------------
// Kernel 3: output projection.  out[8192,1024] = A[8192,1024]*Wout^T + b.
// Same blocking as the QKV GEMM.
// ---------------------------------------------------------------------------
__global__ __launch_bounds__(256)
void out_proj_kernel(const bf16* __restrict__ ab, const bf16* __restrict__ wo,
                     const float* __restrict__ bias, float* __restrict__ out) {
  __shared__ __align__(16) bf16 wbuf[2][64][32];

  const int tid  = threadIdx.x;
  const int lane = tid & 31;
  const int w    = tid >> 5;
  const int rowA = blockIdx.x * 256 + w * 32;
  const int col0 = blockIdx.y * 64;

  auto stage = [&](int kk, int bufi) {
    const int n = tid >> 2, off = (tid & 3) * 8;
    ASYNC_B128(&wbuf[bufi][n][off],
               wo + (size_t)(col0 + n) * HIDDEN + kk + off);
  };

  f32x8 acc[2][4] = {};
  stage(0, 0);
  stage(32, 1);

  const int NSTEP = HIDDEN / 32;
  for (int s = 0; s < NSTEP; ++s) {
    if (s + 1 < NSTEP) { WAIT_ASYNC(0x1); } else { WAIT_ASYNC(0x0); }
    __syncthreads();
    const int bufi = s & 1;
    const int kk   = s * 32;
    bf16x16 a0 = load_a_frag(ab, HIDDEN, rowA,      kk, lane);
    bf16x16 a1 = load_a_frag(ab, HIDDEN, rowA + 16, kk, lane);
    const bf16* wb = &wbuf[bufi][0][0];
#pragma unroll
    for (int t = 0; t < 4; ++t) {
      bf16x16 b = load_bt_frag(wb, 32, t * 16, 0, lane);
      acc[0][t] = wmma_bf16(a0, b, acc[0][t]);
      acc[1][t] = wmma_bf16(a1, b, acc[1][t]);
    }
    __syncthreads();
    if (s + 2 < NSTEP) {
      stage(kk + 64, bufi);
      __builtin_prefetch(ab + (size_t)(rowA + lane) * HIDDEN + kk + 64, 0, 3);
    }
  }

  const int nloc = lane & 15;
  const int hh   = lane >> 4;
#pragma unroll
  for (int g = 0; g < 2; ++g) {
    const int rbase = rowA + g * 16;
#pragma unroll
    for (int t = 0; t < 4; ++t) {
      const int n  = col0 + t * 16 + nloc;
      const float bv = bias[n];
#pragma unroll
      for (int r = 0; r < 8; ++r) {
        const int row = rbase + r + 8 * hh;
        out[(size_t)row * HIDDEN + n] = acc[g][t][r] + bv;
      }
    }
  }
}

// ---------------------------------------------------------------------------
// Host-side launcher.
// ---------------------------------------------------------------------------
extern "C" void kernel_launch(void* const* d_in, const int* in_sizes, int n_in,
                              void* d_out, int out_size, void* d_ws,
                              size_t ws_size, hipStream_t stream) {
  (void)in_sizes; (void)n_in; (void)out_size; (void)ws_size;

  const float* x     = (const float*)d_in[0];   // [4,2048,1024]
  const float* w_qkv = (const float*)d_in[1];   // [3072,1024]
  const float* w_out = (const float*)d_in[2];   // [1024,1024]
  const float* b_out = (const float*)d_in[3];   // [1024]
  float* out = (float*)d_out;

  char* ws = (char*)d_ws;
  size_t off = 0;
  auto walloc = [&](size_t bytes) -> void* {
    void* p = ws + off;
    off += (bytes + 255) & ~(size_t)255;
    return p;
  };
  bf16* xb  = (bf16*)walloc((size_t)TOKENS * HIDDEN * 2);                // 16 MB
  bf16* wqb = (bf16*)walloc((size_t)3 * HIDDEN * HIDDEN * 2);            //  6 MB
  bf16* wob = (bf16*)walloc((size_t)HIDDEN * HIDDEN * 2);                //  2 MB
  bf16* qb  = (bf16*)walloc((size_t)BATCH * HEADS * SEQ * HEAD_DIM * 2); // 16 MB
  bf16* kb  = (bf16*)walloc((size_t)BATCH * HEADS * SEQ * HEAD_DIM * 2); // 16 MB
  bf16* vtb = (bf16*)walloc((size_t)BATCH * HEADS * HEAD_DIM * SEQ * 2); // 16 MB
  bf16* abf = (bf16*)walloc((size_t)TOKENS * HIDDEN * 2);                // 16 MB

  // 0) f32 -> bf16 conversions
  {
    int n4 = TOKENS * HIDDEN / 4;
    cvt_f32_to_bf16_v4<<<(n4 + 255) / 256, 256, 0, stream>>>(x, xb, n4);
    n4 = 3 * HIDDEN * HIDDEN / 4;
    cvt_f32_to_bf16_v4<<<(n4 + 255) / 256, 256, 0, stream>>>(w_qkv, wqb, n4);
    n4 = HIDDEN * HIDDEN / 4;
    cvt_f32_to_bf16_v4<<<(n4 + 255) / 256, 256, 0, stream>>>(w_out, wob, n4);
  }

  // 1) QKV projection: block tile 256(M) x 64(N)
  qkv_gemm_kernel<<<dim3(TOKENS / 256, (3 * HIDDEN) / 64), 256, 0, stream>>>(
      xb, wqb, qb, kb, vtb);

  // 2) attention: block = 128 query rows of one (b,h)
  attn_kernel<<<dim3(SEQ / 128, BATCH * HEADS), 256, 0, stream>>>(qb, kb, vtb,
                                                                  abf);

  // 3) output projection + bias
  out_proj_kernel<<<dim3(TOKENS / 256, HIDDEN / 64), 256, 0, stream>>>(
      abf, wob, b_out, out);
}
